// HeteroGATv2_146028888142
// MI455X (gfx1250) — matrix operand order repaired and checked by reading, not compiled, not run on hardware
//
#include <hip/hip_runtime.h>

// ---------------------------------------------------------------------------
// HeteroGATv2 forward for MI455X (gfx1250, wave32).
//  - Dense GEMMs: v_wmma_f32_16x16x32_f16, W panel staged in LDS as
//    pre-swizzled f16 B-fragments (ds_load_b128), 4 WMMAs per wave per K-step.
//  - Bias slice copied via global_load_async_to_lds_b32 (+ s_wait_asynccnt).
//  - Edge softmax/aggregation via f32 global atomics.
// ---------------------------------------------------------------------------

#define NW   100000
#define NS   50000
#define DIN  128
#define EWW  320000
#define EWS  160000
#define ESW  160000
#define H0   4
#define C0   64
#define D1   256   // H0*C0
#define H1   1
#define C1   64

typedef __attribute__((ext_vector_type(16))) _Float16 v16h;
typedef __attribute__((ext_vector_type(8)))  float    v8f;

// ------------------------- WMMA GEMM: C = A@W + bias ------------------------
// A:[M,K] f32 row-major, W:[K,N] f32 row-major, bias:[N], C:[M,N] f32.
// M%16==0, N%64==0, K%32==0, K<=256.
// Block: 256 threads = 8 waves. Block tile: 128 (M) x 64 (N).
// Each wave: one 16-row strip x 4 16-col tiles (4 accumulators).
__global__ __launch_bounds__(256)
void gat_gemm_wmma(const float* __restrict__ A, const float* __restrict__ W,
                   const float* __restrict__ bias, float* __restrict__ Cmat,
                   int M, int N, int K)
{
    // B fragments: [kt][nt][lane][16 halfs]; max K=256 -> 8 kt * 4 nt = 16384 halfs
    __shared__ __align__(32) _Float16 wfrag[8 * 4 * 32 * 16];
    __shared__ __align__(16) float    sbias[64];

    const int tid   = threadIdx.x;
    const int lane  = tid & 31;
    const int wave  = tid >> 5;
    const int nBase = blockIdx.y * 64;

    // ---- bias slice: async global->LDS copy (ASYNCcnt path) ----------------
    if (tid < 64) {
        unsigned           lds = (unsigned)(size_t)&sbias[tid];
        unsigned long long ga  = (unsigned long long)(size_t)(bias + nBase + tid);
        asm volatile("global_load_async_to_lds_b32 %0, %1, off"
                     :: "v"(lds), "v"(ga) : "memory");
    }

    // ---- stage W panel as pre-swizzled f16 WMMA B-fragments ----------------
    // fragment elem (kt,nt,lane,i) = W[kt*32 + (lane>>4)*16 + i][nBase + nt*16 + (lane&15)]
    const int pairs = (K * 64) >> 1;          // 2 consecutive columns per iter
    for (int p = tid; p < pairs; p += 256) {
        const int    k  = p >> 5;             // panel row 0..K-1
        const int    n0 = (p & 31) << 1;      // panel col (even)
        const float2 w2 = *(const float2*)(W + (size_t)k * N + nBase + n0);
        const int kt = k >> 5, half = (k >> 4) & 1, i = k & 15;
#pragma unroll
        for (int q = 0; q < 2; ++q) {
            const int n  = n0 + q;
            const int nt = n >> 4;
            const int ln = half * 16 + (n & 15);
            wfrag[(((kt * 4 + nt) * 32) + ln) * 16 + i] =
                (_Float16)((q == 0) ? w2.x : w2.y);
        }
    }
    asm volatile("s_wait_asynccnt 0x0" ::: "memory");
    __syncthreads();

    const int Mtiles = M >> 4;
    const int tm     = blockIdx.x * 8 + wave;
    if (tm >= Mtiles) return;                 // after the only barrier: safe

    const int half16 = lane >> 4;
    const int l15    = lane & 15;
    const int m      = tm * 16 + l15;         // A row handled by this lane
    const int aK     = half16 * 16;           // A lane-half K offset
    const float* __restrict__ Arow = A + (size_t)m * K;
    const v16h*  __restrict__ fragArr = (const v16h*)wfrag;

    v8f acc[4] = {{}, {}, {}, {}};
    const int nkt = K >> 5;
    for (int kt = 0; kt < nkt; ++kt) {
        v16h a;
        // A fragment: VGPR v holds K = aK + v*2 (+1)
#pragma unroll
        for (int v = 0; v < 8; ++v) {
            const int    k = (kt << 5) + aK + (v << 1);
            const float2 f = *(const float2*)(Arow + k);
            a[2 * v]     = (_Float16)f.x;
            a[2 * v + 1] = (_Float16)f.y;
        }
#pragma unroll
        for (int nt = 0; nt < 4; ++nt) {
            const v16h b = fragArr[(kt * 4 + nt) * 32 + lane];   // 2x ds_load_b128
            acc[nt] = __builtin_amdgcn_wmma_f32_16x16x32_f16(
                /*neg_a=*/false, a, /*neg_b=*/false, b,
                /*c_mod=*/(short)0, acc[nt], /*reuse_a=*/false, /*reuse_b=*/false);
        }
    }

    const int rbase = tm * 16 + half16 * 8;   // C rows: M = rbase + r
#pragma unroll
    for (int nt = 0; nt < 4; ++nt) {
        const int   n  = nBase + nt * 16 + l15;
        const float bn = sbias[nt * 16 + l15];
#pragma unroll
        for (int r = 0; r < 8; ++r)
            Cmat[(size_t)(rbase + r) * N + n] = acc[nt][r] + bn;
    }
}

// ----------------------------- helper kernels -------------------------------

// Monotone float<->uint mapping for atomic max on f32 (handles negatives).
__device__ __forceinline__ unsigned fenc(float f) {
    unsigned u = __float_as_uint(f);
    return (u & 0x80000000u) ? ~u : (u | 0x80000000u);
}
__device__ __forceinline__ float fdec(unsigned e) {
    unsigned u = (e & 0x80000000u) ? (e ^ 0x80000000u) : ~e;
    return __uint_as_float(u);
}
#define FENC_NEG_INF 0x007FFFFFu  // fenc(-inf)

__global__ void fill_u32_kernel(unsigned* __restrict__ p, unsigned v, int n) {
    int i = blockIdx.x * blockDim.x + threadIdx.x;
    if (i < n) p[i] = v;
}

// out[i][j] = b1[j] + (b2 ? b2[j] : 0)   (pre-load conv biases into accumulator)
__global__ void init_bias_kernel(float* __restrict__ out, const float* __restrict__ b1,
                                 const float* __restrict__ b2, int rows, int cols) {
    int i = blockIdx.x * blockDim.x + threadIdx.x;
    if (i >= rows * cols) return;
    int j = i % cols;
    out[i] = b1[j] + (b2 ? b2[j] : 0.0f);
}

__global__ void relu_kernel(float* __restrict__ x, int n) {
    int i = blockIdx.x * blockDim.x + threadIdx.x;
    if (i < n) x[i] = fmaxf(x[i], 0.0f);
}

// logits[e,h] = sum_c att[h,c] * leaky_relu(xl[src,h,c] + xr[dst,h,c], 0.2)
// + atomic segment-max over dst.
__global__ void edge_logits_kernel(const float* __restrict__ xl, const float* __restrict__ xr,
                                   const int* __restrict__ src, const int* __restrict__ dst,
                                   const float* __restrict__ att, float* __restrict__ logit,
                                   unsigned* __restrict__ segmax, int E, int H, int C) {
    int idx = blockIdx.x * blockDim.x + threadIdx.x;
    if (idx >= E * H) return;
    int e = idx / H, h = idx - e * H;
    int s = src[e], d = dst[e];
    const float* __restrict__ pl = xl + (size_t)s * H * C + h * C;
    const float* __restrict__ pr = xr + (size_t)d * H * C + h * C;
    const float* __restrict__ pa = att + h * C;
    float acc = 0.0f;
    for (int c = 0; c < C; ++c) {
        float v = pl[c] + pr[c];
        v = (v > 0.0f) ? v : 0.2f * v;
        acc += pa[c] * v;
    }
    logit[idx] = acc;
    atomicMax(segmax + (size_t)d * H + h, fenc(acc));
}

// ex[e,h] = exp(logit - segmax[dst,h]); denom[dst,h] += ex (atomic).
__global__ void edge_exp_kernel(const int* __restrict__ dst, float* __restrict__ logit_ex,
                                const unsigned* __restrict__ segmax, float* __restrict__ denom,
                                int E, int H) {
    int idx = blockIdx.x * blockDim.x + threadIdx.x;
    if (idx >= E * H) return;
    int e = idx / H, h = idx - e * H;
    int d = dst[e];
    float m = fdec(segmax[(size_t)d * H + h]);
    if (!(m > -3.0e38f && m < 3.0e38f)) m = 0.0f;   // mirror isfinite guard
    float ex = expf(logit_ex[idx] - m);
    logit_ex[idx] = ex;
    atomicAdd(denom + (size_t)d * H + h, ex);
}

// out[dst,h,c] += (ex[e,h] / (denom[dst,h]+1e-16)) * xl[src,h,c]  (atomic)
__global__ void edge_scatter_kernel(const float* __restrict__ xl, const int* __restrict__ src,
                                    const int* __restrict__ dst, const float* __restrict__ ex,
                                    const float* __restrict__ denom, float* __restrict__ out,
                                    int E, int H, int C) {
    int idx = blockIdx.x * blockDim.x + threadIdx.x;
    int total = E * H * C;
    if (idx >= total) return;
    int c  = idx % C;
    int eh = idx / C;
    int h  = eh % H;
    int e  = eh / H;
    int s = src[e], d = dst[e];
    float alpha = ex[eh] / (denom[(size_t)d * H + h] + 1e-16f);
    atomicAdd(out + (size_t)d * H * C + h * C + c,
              alpha * xl[(size_t)s * H * C + h * C + c]);
}

// ------------------------------- host side ----------------------------------

static inline int cdiv(long long a, int b) { return (int)((a + b - 1) / b); }

// One GATv2 conv for one edge type, accumulating into out_accum (biases pre-loaded).
static void run_gat_type(const float* Xs, int Ns, const float* Xd, int Nd, int Din,
                         const int* src, const int* dst, int E,
                         const float* Wl, const float* bl,
                         const float* Wr, const float* br,
                         const float* att, int H, int Cc,
                         float* xl, float* xr, float* ebuf,
                         float* denom, unsigned* segmax,
                         float* out_accum, hipStream_t stream)
{
    const int Ncols = H * Cc;
    {
        dim3 g(cdiv(Ns / 16, 8), Ncols / 64);
        gat_gemm_wmma<<<g, 256, 0, stream>>>(Xs, Wl, bl, xl, Ns, Ncols, Din);
    }
    {
        dim3 g(cdiv(Nd / 16, 8), Ncols / 64);
        gat_gemm_wmma<<<g, 256, 0, stream>>>(Xd, Wr, br, xr, Nd, Ncols, Din);
    }

    const int segn = Nd * H;
    fill_u32_kernel<<<cdiv(segn, 256), 256, 0, stream>>>(segmax, FENC_NEG_INF, segn);
    fill_u32_kernel<<<cdiv(segn, 256), 256, 0, stream>>>((unsigned*)denom, 0u, segn);

    const int eh = E * H;
    edge_logits_kernel<<<cdiv(eh, 256), 256, 0, stream>>>(xl, xr, src, dst, att, ebuf,
                                                          segmax, E, H, Cc);
    edge_exp_kernel<<<cdiv(eh, 256), 256, 0, stream>>>(dst, ebuf, segmax, denom, E, H);
    const long long ehc = (long long)E * H * Cc;
    edge_scatter_kernel<<<cdiv(ehc, 256), 256, 0, stream>>>(xl, src, dst, ebuf, denom,
                                                            out_accum, E, H, Cc);
}

extern "C" void kernel_launch(void* const* d_in, const int* in_sizes, int n_in,
                              void* d_out, int out_size, void* d_ws, size_t ws_size,
                              hipStream_t stream)
{
    (void)in_sizes; (void)n_in; (void)out_size; (void)ws_size;

    const float* x_word  = (const float*)d_in[0];
    const float* x_sense = (const float*)d_in[1];

    // Params flattened in source (insertion) order:
    // groups g: 0=l0.ww 1=l0.ws 2=l0.sw 3=l1.ww 4=l1.ws 5=l1.sw
    // leaves l: 0=Wl 1=bl 2=Wr 3=br 4=att 5=bias
    const float* P[6][6];
    for (int g = 0; g < 6; ++g)
        for (int l = 0; l < 6; ++l)
            P[g][l] = (const float*)d_in[2 + g * 6 + l];
    enum { WL = 0, BL = 1, WR = 2, BR = 3, ATT = 4, BIAS = 5 };
    enum { L0WW = 0, L0WS = 1, L0SW = 2, L1WW = 3, L1WS = 4, L1SW = 5 };

    const int* ei_ww = (const int*)d_in[38];  // [2,EWW]: row0 src, row1 dst
    const int* ei_ws = (const int*)d_in[39];
    const int* ei_sw = (const int*)d_in[40];

    // ------------------- workspace carving (256B aligned) -------------------
    size_t off = 0;
    char* base = (char*)d_ws;
    auto carve = [&](size_t bytes) -> void* {
        void* p = base + off;
        off += (bytes + 255) & ~(size_t)255;
        return p;
    };
    float*    xl      = (float*)carve((size_t)NW * D1 * 4);   // src-projected feats
    float*    xr      = (float*)carve((size_t)NW * D1 * 4);   // dst-projected feats
    float*    h_word  = (float*)carve((size_t)NW * D1 * 4);
    float*    h_sense = (float*)carve((size_t)NS * D1 * 4);
    float*    ebuf    = (float*)carve((size_t)EWW * H0 * 4);  // per-edge logits/exp
    float*    denom   = (float*)carve((size_t)NW * H0 * 4);
    unsigned* segmax  = (unsigned*)carve((size_t)NW * H0 * 4);

    // ------------------------------ layer 0 ---------------------------------
    init_bias_kernel<<<cdiv((long long)NW * D1, 256), 256, 0, stream>>>(
        h_word, P[L0WW][BIAS], P[L0SW][BIAS], NW, D1);
    init_bias_kernel<<<cdiv((long long)NS * D1, 256), 256, 0, stream>>>(
        h_sense, P[L0WS][BIAS], nullptr, NS, D1);

    // word<-word
    run_gat_type(x_word, NW, x_word, NW, DIN, ei_ww, ei_ww + EWW, EWW,
                 P[L0WW][WL], P[L0WW][BL], P[L0WW][WR], P[L0WW][BR], P[L0WW][ATT],
                 H0, C0, xl, xr, ebuf, denom, segmax, h_word, stream);
    // word<-sense
    run_gat_type(x_sense, NS, x_word, NW, DIN, ei_sw, ei_sw + ESW, ESW,
                 P[L0SW][WL], P[L0SW][BL], P[L0SW][WR], P[L0SW][BR], P[L0SW][ATT],
                 H0, C0, xl, xr, ebuf, denom, segmax, h_word, stream);
    // sense<-word
    run_gat_type(x_word, NW, x_sense, NS, DIN, ei_ws, ei_ws + EWS, EWS,
                 P[L0WS][WL], P[L0WS][BL], P[L0WS][WR], P[L0WS][BR], P[L0WS][ATT],
                 H0, C0, xl, xr, ebuf, denom, segmax, h_sense, stream);

    relu_kernel<<<cdiv((long long)NW * D1, 256), 256, 0, stream>>>(h_word, NW * D1);
    relu_kernel<<<cdiv((long long)NS * D1, 256), 256, 0, stream>>>(h_sense, NS * D1);

    // ------------------------------ layer 1 ---------------------------------
    float* o_word  = (float*)d_out;                 // [NW,64]
    float* o_sense = o_word + (size_t)NW * C1;      // [NS,64]

    init_bias_kernel<<<cdiv((long long)NW * C1, 256), 256, 0, stream>>>(
        o_word, P[L1WW][BIAS], P[L1SW][BIAS], NW, C1);
    init_bias_kernel<<<cdiv((long long)NS * C1, 256), 256, 0, stream>>>(
        o_sense, P[L1WS][BIAS], nullptr, NS, C1);

    run_gat_type(h_word, NW, h_word, NW, D1, ei_ww, ei_ww + EWW, EWW,
                 P[L1WW][WL], P[L1WW][BL], P[L1WW][WR], P[L1WW][BR], P[L1WW][ATT],
                 H1, C1, xl, xr, ebuf, denom, segmax, o_word, stream);
    run_gat_type(h_sense, NS, h_word, NW, D1, ei_sw, ei_sw + ESW, ESW,
                 P[L1SW][WL], P[L1SW][BL], P[L1SW][WR], P[L1SW][BR], P[L1SW][ATT],
                 H1, C1, xl, xr, ebuf, denom, segmax, o_word, stream);
    run_gat_type(h_word, NW, h_sense, NS, D1, ei_ws, ei_ws + EWS, EWS,
                 P[L1WS][WL], P[L1WS][BL], P[L1WS][WR], P[L1WS][BR], P[L1WS][ATT],
                 H1, C1, xl, xr, ebuf, denom, segmax, o_sense, stream);
}
